// TensorProductSelfInteraction_27573690040903
// MI455X (gfx1250) — compile-verified, matrix operand order checked by reading, not run on hardware
//
#include <hip/hip_runtime.h>
#include <cstdint>

// CDNA5 (gfx1250) wave32 WMMA implementation of the e3nn
// FullyConnectedTensorProduct with scalar Z irreps.
//
// o[n,w,i] = alpha * sum_{u,v} x[n,u,i] * Z[n,v] * W[u,v,w]
//  == GEMM with A[(n,i), u*16+v] = x[n,u,i]*Z[n,v] (built on the fly in LDS,f16)
//     and  B[u*16+v, w]          = alpha*W[u,v,w]  (pre-converted to f16 in ws)
//
// v2: 32 nodes per block + row-tile register blocking (RB row tiles per wave)
//     so each B fragment (global/L2) feeds RB WMMAs -> 2-3x less L2 traffic.

typedef __attribute__((ext_vector_type(16))) _Float16 v16h;
typedef __attribute__((ext_vector_type(8)))  float    v8f;

union Frag16 { uint32_t u[8]; v16h h; };
union Pack16 { _Float16 h[16]; uint4 q[2]; };

// ---------------------------------------------------------------------------
// Weight pre-conversion: W[u,v,w] (f32, u-major) -> Bt[w*K + u*16+v] (f16),
// alpha folded in. Runs every launch (deterministic), ~344K elements total.
// ---------------------------------------------------------------------------
__global__ void wcvt_kernel(const float* __restrict__ W, _Float16* __restrict__ Bt,
                            int m_out, int K, float alpha) {
  const int total = K * m_out;
  for (int idx = blockIdx.x * blockDim.x + threadIdx.x; idx < total;
       idx += gridDim.x * blockDim.x) {
    const int w = idx / K;
    const int k = idx - w * K;                 // k = u*16 + v
    Bt[idx] = (_Float16)(W[k * m_out + w] * alpha);
  }
}

// ---------------------------------------------------------------------------
// Main tensor-product GEMM kernel (one instantiation per l-block).
//   MIN/MOUT : multiplicities (K = MIN*16)
//   D        : 2l+1
//   OFF_IN/OFF_OUT : column offsets into the 480-wide irreps tensor
//   NODES    : nodes per block (32)
//   RB       : row tiles (of 16 rows) per wave; one wave = 16 cols x RB*16 rows
// ---------------------------------------------------------------------------
template<int MIN, int MOUT, int D, int OFF_IN, int OFF_OUT, int NODES, int RB>
__global__ __launch_bounds__((MOUT / 16) * (NODES * D / 16 / RB) * 32)
void tp_wmma_kernel(const float* __restrict__ x, const float* __restrict__ z,
                    const _Float16* __restrict__ Bt, float* __restrict__ out) {
  constexpr int K        = MIN * 16;
  constexpr int ROWS     = NODES * D;            // (node,i) rows per block
  constexpr int ROWTILES = ROWS / 16;
  constexpr int CT       = MOUT / 16;            // column tiles
  constexpr int NWAVES   = CT * (ROWTILES / RB);
  constexpr int NTHR     = NWAVES * 32;
  constexpr int KCHUNK   = (ROWS <= 32) ? 512 : (ROWS <= 96 ? 256 : 128);
  constexpr int KLD      = KCHUNK + 8;           // pad 16B: de-conflict LDS banks
  static_assert(ROWTILES % RB == 0, "row tiles must split evenly across waves");
  static_assert(K % KCHUNK == 0, "K must be a multiple of KCHUNK");
  static_assert((size_t)ROWS * KLD * 2 < 56 * 1024, "A tile too large for LDS");

  __shared__ _Float16 Ash[ROWS][KLD];
  __shared__ float    Zs[NODES * 16];

  const int nodeBase = blockIdx.x * NODES;
  const int tid  = threadIdx.x;
  const int wave = tid >> 5;
  const int lane = tid & 31;
  const int nl16 = lane & 15;   // M row (A frag) / N col (B,C frags)
  const int half = lane >> 4;   // K-half select for 16-bit A/B fragments

  // Stage Z tile (NODES x 16) into LDS.
  for (int idx = tid; idx < NODES * 16; idx += NTHR) {
    Zs[idx] = z[(nodeBase + (idx >> 4)) * 16 + (idx & 15)];
  }
  __syncthreads();

  const int colTile  = wave % CT;
  const int rowGroup = wave / CT;      // owns row tiles [rowGroup*RB, ...+RB)

  v8f acc[RB];
#pragma unroll
  for (int rb = 0; rb < RB; ++rb) acc[rb] = (v8f){};

  for (int kk0 = 0; kk0 < K; kk0 += KCHUNK) {
    // -------- build Kronecker A chunk in LDS (f16) --------
    constexpr int GROUPS = KCHUNK / 16;          // 16 k's (one u, all v) per task
    constexpr int TASKS  = ROWS * GROUPS;
    for (int idx = tid; idx < TASKS; idx += NTHR) {
      const int r  = idx / GROUPS;
      const int g  = idx - r * GROUPS;
      const int u  = (kk0 >> 4) + g;
      const int nl = r / D;
      const int i  = r - nl * D;
      const float xv = x[(nodeBase + nl) * 480 + OFF_IN + u * D + i];
      const float* zr = &Zs[nl * 16];
      Pack16 p;
#pragma unroll
      for (int v = 0; v < 16; ++v) p.h[v] = (_Float16)(xv * zr[v]);
      *(uint4*)&Ash[r][g * 16]     = p.q[0];
      *(uint4*)&Ash[r][g * 16 + 8] = p.q[1];
    }
    __syncthreads();

    // -------- WMMA accumulate over this K chunk --------
    const _Float16* __restrict__ Bcol =
        Bt + (size_t)(colTile * 16 + nl16) * K + kk0;
    const _Float16* __restrict__ Arow0 = &Ash[rowGroup * RB * 16 + nl16][0];
#pragma unroll 4
    for (int kk = 0; kk < KCHUNK; kk += 32) {
      // One B fragment feeds RB WMMAs (register row-blocking).
      Frag16 b;
#pragma unroll
      for (int j = 0; j < 8; ++j) {
        // 16-bit fragment K map: VGPR j<4 -> K={2j,2j+1}+8*half,
        //                        j>=4   -> K=16+{2(j-4),..}+8*half
        const int kb = (j < 4) ? (8 * half + 2 * j)
                               : (16 + 8 * half + 2 * (j - 4));
        b.u[j] = *(const uint32_t*)&Bcol[kk + kb];
      }
#pragma unroll
      for (int rb = 0; rb < RB; ++rb) {
        Frag16 a;
#pragma unroll
        for (int j = 0; j < 8; ++j) {
          const int kb = (j < 4) ? (8 * half + 2 * j)
                                 : (16 + 8 * half + 2 * (j - 4));
          a.u[j] = *(const uint32_t*)&Arow0[rb * 16 * KLD + kk + kb];
        }
        acc[rb] = __builtin_amdgcn_wmma_f32_16x16x32_f16(
            /*neg_a=*/false, a.h, /*neg_b=*/false, b.h,
            /*c_mod=*/(short)0, acc[rb], /*reuse_a=*/false, /*reuse_b=*/false);
      }
    }
    __syncthreads();
  }

  // -------- scatter C (f32): VGPR j holds row M=j+8*half, col=lane%16 --------
#pragma unroll
  for (int rb = 0; rb < RB; ++rb) {
#pragma unroll
    for (int j = 0; j < 8; ++j) {
      const int R  = (rowGroup * RB + rb) * 16 + j + 8 * half;
      const int nl = R / D;
      const int i  = R - nl * D;
      const int w  = colTile * 16 + nl16;
      out[(nodeBase + nl) * 480 + OFF_OUT + w * D + i] = acc[rb][j];
    }
  }
}

// ---------------------------------------------------------------------------
extern "C" void kernel_launch(void* const* d_in, const int* in_sizes, int n_in,
                              void* d_out, int out_size, void* d_ws, size_t ws_size,
                              hipStream_t stream) {
  const float* x  = (const float*)d_in[0];   // [N,480]
  const float* z  = (const float*)d_in[1];   // [N,16]
  const float* W0 = (const float*)d_in[2];   // [128,16,128]
  const float* W1 = (const float*)d_in[3];   // [64,16,64]
  const float* W2 = (const float*)d_in[4];   // [32,16,32]
  float* out = (float*)d_out;

  _Float16* Bt0 = (_Float16*)d_ws;           // 2048*128 f16 = 512 KB
  _Float16* Bt1 = Bt0 + 2048 * 128;          // 1024*64  f16 = 128 KB
  _Float16* Bt2 = Bt1 + 1024 * 64;           // 512*32   f16 =  32 KB

  // alpha = 1/sqrt(m_in * 16), folded into the f16 weights.
  wcvt_kernel<<<512, 256, 0, stream>>>(W0, Bt0, 128, 2048, 0.02209708691207961f);
  wcvt_kernel<<<128, 256, 0, stream>>>(W1, Bt1,  64, 1024, 0.03125f);
  wcvt_kernel<<< 32, 256, 0, stream>>>(W2, Bt2,  32,  512, 0.04419417382415922f);

  const int N = in_sizes[0] / 480;           // 100000
  const int nblocks = N / 32;                // 3125 (exact)

  // l=0: 32 rows -> 2 row tiles, RB=2 -> 8 waves (256 thr), 1 B-frag : 2 WMMA
  tp_wmma_kernel<128, 128, 1,   0,   0, 32, 2>
      <<<nblocks, 256, 0, stream>>>(x, z, Bt0, out);
  // l=1: 96 rows -> 6 row tiles, RB=3 -> 8 waves (256 thr), 1 B-frag : 3 WMMA
  tp_wmma_kernel< 64,  64, 3, 128, 128, 32, 3>
      <<<nblocks, 256, 0, stream>>>(x, z, Bt1, out);
  // l=2: 160 rows -> 10 row tiles, RB=2 -> 10 waves (320 thr), 1 B-frag : 2 WMMA
  tp_wmma_kernel< 32,  32, 5, 320, 320, 32, 2>
      <<<nblocks, 320, 0, stream>>>(x, z, Bt2, out);
}
